// MSDeformAttn_21680994910425
// MI455X (gfx1250) — matrix operand order repaired and checked by reading, not compiled, not run on hardware
//
#include <hip/hip_runtime.h>
#include <hip/hip_bf16.h>

typedef __attribute__((ext_vector_type(16))) _Float16 v16h;
typedef __attribute__((ext_vector_type(8)))  _Float16 v8h;
typedef __attribute__((ext_vector_type(8)))  float    v8f;

#define B_  4
#define Q_  8192
#define D_  256
#define H_  8
#define L_  4
#define P_  4
#define HD_ 32
#define M_  (B_ * Q_)            // 32768 rows
#define N1_ 384                  // 256 offset cols + 128 attn cols
#define N2_ 256
#define K_  256

// ---------------------------------------------------------------------------
// Elementwise fp32 -> f16 conversion (query)
// ---------------------------------------------------------------------------
__global__ __launch_bounds__(256) void cvt_f32_f16(const float* __restrict__ src,
                                                   _Float16* __restrict__ dst, int n) {
    int i = blockIdx.x * 256 + threadIdx.x;
    if (i < n) dst[i] = (_Float16)src[i];
}

// ---------------------------------------------------------------------------
// Pack weights: Bt1 = [W_off | W_attn]^T (384 x 256, f16), b1 = [b_off|b_attn],
//               Bt2 = W_out^T (256 x 256, f16)
// ---------------------------------------------------------------------------
__global__ __launch_bounds__(256) void pack_weights(const float* __restrict__ W_off,
                                                    const float* __restrict__ W_attn,
                                                    const float* __restrict__ b_off,
                                                    const float* __restrict__ b_attn,
                                                    const float* __restrict__ W_out,
                                                    _Float16* __restrict__ Bt1,
                                                    float* __restrict__ b1,
                                                    _Float16* __restrict__ Bt2) {
    int id = blockIdx.x * 256 + threadIdx.x;
    if (id < N1_ * K_) {
        int n = id >> 8, k = id & 255;
        float v = (n < 256) ? W_off[k * 256 + n] : W_attn[k * 128 + (n - 256)];
        Bt1[n * K_ + k] = (_Float16)v;
        if (k == 0) b1[n] = (n < 256) ? b_off[n] : b_attn[n - 256];
    } else {
        int id2 = id - N1_ * K_;
        if (id2 < N2_ * K_) {
            int n = id2 >> 8, k = id2 & 255;
            Bt2[n * K_ + k] = (_Float16)W_out[k * 256 + n];
        }
    }
}

// ---------------------------------------------------------------------------
// WMMA fragment helpers (CDNA5 16x16x32 f16 layouts, wave32)
// A (16x32, row-major MxK): lane<16 -> M=lane,   K in {k0..k0+7, k0+16..k0+23}
//                           lane>=16-> M=lane-16,K in {k0+8..15, k0+24..31}
// B via Bt (NxK row-major): lanes0-15 N=lane, K in {k0..k0+7, k0+16..k0+23};
//                           lanes16-31 N=lane-16, K in {k0+8..15, k0+24..31}
// ---------------------------------------------------------------------------
__device__ inline v16h load_a_frag(const _Float16* __restrict__ A, int row, int k0, int lane) {
    union { v16h v; v8h h[2]; } a;
    const _Float16* p = A + (size_t)row * K_ + k0 + ((lane >> 4) << 3);
    a.h[0] = *(const v8h*)(p);
    a.h[1] = *(const v8h*)(p + 16);
    return a.v;
}
__device__ inline v16h load_b_frag(const _Float16* __restrict__ Bt, int n, int k0, int lane) {
    union { v16h v; v8h h[2]; } b;
    const _Float16* p = Bt + (size_t)n * K_ + k0 + ((lane >> 4) << 3);
    b.h[0] = *(const v8h*)(p);
    b.h[1] = *(const v8h*)(p + 16);
    return b.v;
}

// ---------------------------------------------------------------------------
// GEMM1: [M x 384] = query_f16 @ [W_off|W_attn] ; fused bias + tanh (cols<256)
// and per-head softmax over 16 logits (cols>=256). Each wave: 16x64 tile.
// ---------------------------------------------------------------------------
__global__ __launch_bounds__(256) void gemm1_fused(const _Float16* __restrict__ qh,
                                                   const _Float16* __restrict__ Bt1,
                                                   const float* __restrict__ b1,
                                                   float* __restrict__ off_out,   // M x 256 (tanh)
                                                   float* __restrict__ attn_out)  // M x 128 (softmax)
{
    const int TN = N1_ / 64;                       // 6 wave-tiles along N
    int wid  = (blockIdx.x * 256 + threadIdx.x) >> 5;
    int lane = threadIdx.x & 31;
    int tm = wid / TN, tn = wid % TN;
    if (tm * 16 >= M_) return;

    v8f acc[4] = {};
    int rowA = tm * 16 + (lane & 15);
    int nB   = tn * 64 + (lane & 15);

    #pragma unroll
    for (int k0 = 0; k0 < K_; k0 += 32) {
        v16h a = load_a_frag(qh, rowA, k0, lane);
        #pragma unroll
        for (int j = 0; j < 4; ++j) {
            v16h b = load_b_frag(Bt1, nB + j * 16, k0, lane);
            acc[j] = __builtin_amdgcn_wmma_f32_16x16x32_f16(
                false, a, false, b, (short)0, acc[j], false, false);
        }
    }

    int rhalf = (lane >> 4) << 3;                  // +8 rows for upper half-wave
    #pragma unroll
    for (int j = 0; j < 4; ++j) {
        int col0 = tn * 64 + j * 16;
        int col  = col0 + (lane & 15);
        float bv = b1[col];
        if (col0 < 256) {                          // offset region -> tanh
            #pragma unroll
            for (int i = 0; i < 8; ++i) {
                int row = tm * 16 + i + rhalf;
                off_out[(size_t)row * 256 + col] = tanhf(acc[j][i] + bv);
            }
        } else {                                   // logit region -> softmax over 16 cols
            #pragma unroll
            for (int i = 0; i < 8; ++i) {
                int row = tm * 16 + i + rhalf;
                float v = acc[j][i] + bv;
                float m = v;
                for (int o = 1; o < 16; o <<= 1) m = fmaxf(m, __shfl_xor(m, o, 32));
                float e = __expf(v - m);
                float s = e;
                for (int o = 1; o < 16; o <<= 1) s += __shfl_xor(s, o, 32);
                attn_out[(size_t)row * 128 + (col - 256)] = e / s;
            }
        }
    }
}

// ---------------------------------------------------------------------------
// Bilinear sampling + attention aggregation. One wave per (b,q,h); lane = HD
// channel. Feature pyramid (85 MB) is L2-resident on MI455X (192 MB L2).
// ---------------------------------------------------------------------------
template <int HW>
__device__ inline float sample_level(const float* __restrict__ plane, float gx, float gy,
                                     const float* __restrict__ off8,
                                     const float* __restrict__ att4) {
    float acc = 0.f;
    #pragma unroll
    for (int p = 0; p < 4; ++p) {
        float x = (gx + off8[p * 2 + 0] + 1.f) * 0.5f * (float)(HW - 1);
        float y = (gy + off8[p * 2 + 1] + 1.f) * 0.5f * (float)(HW - 1);
        float x0f = floorf(x), y0f = floorf(y);
        int x0 = (int)x0f, y0 = (int)y0f;
        float fx = x - x0f, fy = y - y0f;
        float w00 = (1.f - fx) * (1.f - fy), w10 = fx * (1.f - fy);
        float w01 = (1.f - fx) * fy,         w11 = fx * fy;
        bool vx0 = (x0 >= 0) & (x0 < HW), vx1 = (x0 + 1 >= 0) & (x0 + 1 < HW);
        bool vy0 = (y0 >= 0) & (y0 < HW), vy1 = (y0 + 1 >= 0) & (y0 + 1 < HW);
        float v = 0.f;
        if (vx0 & vy0) v += w00 * plane[y0 * HW + x0];
        if (vx1 & vy0) v += w10 * plane[y0 * HW + x0 + 1];
        if (vx0 & vy1) v += w01 * plane[(y0 + 1) * HW + x0];
        if (vx1 & vy1) v += w11 * plane[(y0 + 1) * HW + x0 + 1];
        acc += att4[p] * v;
    }
    return acc;
}

__global__ __launch_bounds__(256) void msdeform_sample(const float* __restrict__ feat0,
                                                       const float* __restrict__ feat1,
                                                       const float* __restrict__ feat2,
                                                       const float* __restrict__ feat3,
                                                       const float* __restrict__ refp,
                                                       const float* __restrict__ offs,
                                                       const float* __restrict__ attn,
                                                       _Float16* __restrict__ aggh) {
    int gid  = blockIdx.x * 256 + threadIdx.x;
    int wave = gid >> 5;
    int lane = gid & 31;               // channel within head
    int row  = wave >> 3;              // b*Q + q
    int h    = wave & 7;
    if (row >= M_) return;
    int b = row >> 13;                 // Q = 8192

    float gx = refp[row * 2 + 0] * 2.f - 1.f;
    float gy = refp[row * 2 + 1] * 2.f - 1.f;
    const float* offH = offs + (size_t)row * 256 + h * 32;   // (L,P,2) for this head
    const float* attH = attn + (size_t)row * 128 + h * 16;   // (L,P)
    size_t ch = (size_t)(b * D_ + h * HD_ + lane);

    float acc = 0.f;
    acc += sample_level<128>(feat0 + ch * (128 * 128), gx, gy, offH + 0,  attH + 0);
    acc += sample_level<64> (feat1 + ch * (64 * 64),   gx, gy, offH + 8,  attH + 4);
    acc += sample_level<32> (feat2 + ch * (32 * 32),   gx, gy, offH + 16, attH + 8);
    acc += sample_level<16> (feat3 + ch * (16 * 16),   gx, gy, offH + 24, attH + 12);

    aggh[(size_t)row * 256 + h * 32 + lane] = (_Float16)acc;
}

// ---------------------------------------------------------------------------
// GEMM2: out[M x 256] = agg_f16 @ W_out + b_out
// ---------------------------------------------------------------------------
__global__ __launch_bounds__(256) void gemm2_out(const _Float16* __restrict__ aggh,
                                                 const _Float16* __restrict__ Bt2,
                                                 const float* __restrict__ b_out,
                                                 float* __restrict__ out) {
    const int TN = N2_ / 64;                       // 4
    int wid  = (blockIdx.x * 256 + threadIdx.x) >> 5;
    int lane = threadIdx.x & 31;
    int tm = wid / TN, tn = wid % TN;
    if (tm * 16 >= M_) return;

    v8f acc[4] = {};
    int rowA = tm * 16 + (lane & 15);
    int nB   = tn * 64 + (lane & 15);

    #pragma unroll
    for (int k0 = 0; k0 < K_; k0 += 32) {
        v16h a = load_a_frag(aggh, rowA, k0, lane);
        #pragma unroll
        for (int j = 0; j < 4; ++j) {
            v16h b = load_b_frag(Bt2, nB + j * 16, k0, lane);
            acc[j] = __builtin_amdgcn_wmma_f32_16x16x32_f16(
                false, a, false, b, (short)0, acc[j], false, false);
        }
    }

    int rhalf = (lane >> 4) << 3;
    #pragma unroll
    for (int j = 0; j < 4; ++j) {
        int col = tn * 64 + j * 16 + (lane & 15);
        float bv = b_out[col];
        #pragma unroll
        for (int i = 0; i < 8; ++i) {
            int row = tm * 16 + i + rhalf;
            out[(size_t)row * 256 + col] = acc[j][i] + bv;
        }
    }
}

// ---------------------------------------------------------------------------
extern "C" void kernel_launch(void* const* d_in, const int* in_sizes, int n_in,
                              void* d_out, int out_size, void* d_ws, size_t ws_size,
                              hipStream_t stream) {
    const float* query = (const float*)d_in[0];
    const float* feat0 = (const float*)d_in[1];
    const float* feat1 = (const float*)d_in[2];
    const float* feat2 = (const float*)d_in[3];
    const float* feat3 = (const float*)d_in[4];
    const float* refp  = (const float*)d_in[5];
    const float* W_off = (const float*)d_in[6];
    const float* b_off = (const float*)d_in[7];
    const float* W_attn= (const float*)d_in[8];
    const float* b_attn= (const float*)d_in[9];
    const float* W_out = (const float*)d_in[10];
    const float* b_out = (const float*)d_in[11];

    float* out  = (float*)d_out;                 // (B,Q,D) = 8,388,608 floats
    float* attn = out + (size_t)M_ * D_;         // (B,Q,H,L,P) = 4,194,304 floats

    char* ws = (char*)d_ws;
    _Float16* qh   = (_Float16*)(ws);                        // 16,777,216 B
    _Float16* aggh = (_Float16*)(ws + (size_t)16777216);     // 16,777,216 B
    _Float16* Bt1  = (_Float16*)(ws + (size_t)33554432);     //    196,608 B
    _Float16* Bt2  = (_Float16*)(ws + (size_t)33751040);     //    131,072 B
    float*    b1   = (float*)   (ws + (size_t)33882112);     //      1,536 B
    float*    offs = (float*)   (ws + (size_t)33883648);     // 33,554,432 B (tanh'd offsets, M x 256)

    // 1) query -> f16
    cvt_f32_f16<<<(M_ * K_) / 256, 256, 0, stream>>>(query, qh, M_ * K_);

    // 2) pack transposed f16 weights + combined bias
    pack_weights<<<(N1_ * K_ + N2_ * K_ + 255) / 256, 256, 0, stream>>>(
        W_off, W_attn, b_off, b_attn, W_out, Bt1, b1, Bt2);

    // 3) GEMM1 + tanh + softmax (attn written directly to its output slot)
    {
        int waves = (M_ / 16) * (N1_ / 64);      // 12288
        gemm1_fused<<<waves / 8, 256, 0, stream>>>(qh, Bt1, b1, offs, attn);
    }

    // 4) deformable bilinear sampling + aggregation (L2-resident pyramid)
    {
        int waves = M_ * H_;                     // 262144
        msdeform_sample<<<waves / 8, 256, 0, stream>>>(
            feat0, feat1, feat2, feat3, refp, offs, attn, aggh);
    }

    // 5) output projection
    {
        int waves = (M_ / 16) * (N2_ / 64);      // 8192
        gemm2_out<<<waves / 8, 256, 0, stream>>>(aggh, Bt2, b_out, out);
    }
}
